// RSSM_43636867727771
// MI455X (gfx1250) — matrix operand order repaired and checked by reading, not compiled, not run on hardware
//
#include <hip/hip_runtime.h>
#include <hip/hip_bf16.h>

// ---------------------------------------------------------------------------
// RSSM scan for MI455X (gfx1250, wave32, WMMA).
// Persistent kernel: 64 WGs x 256 threads, device-wide spin barriers between
// dependency stages. Weights converted once/launch to bf16, transposed [N][K].
// Matrix math: v_wmma_f32_16x16x32_bf16, fp32 accumulate. Deterministic:
// no fp32 atomics; split-K writes disjoint partial buffers combined in the
// RMS-norm pass. Prior head (3 GEMMs) batched over all B*T rows after scan.
// Row masking (is_first) applied at the C-store (commutes with matmul), so
// the inner K loops are branchless. Workspace requirement: ~66 MB.
// ---------------------------------------------------------------------------

typedef __attribute__((ext_vector_type(16))) __bf16 bf16x16;
typedef __attribute__((ext_vector_type(8)))  float  f32x8;
typedef __attribute__((ext_vector_type(4)))  unsigned int u32x4;

union V16 { bf16x16 v; u32x4 q[2]; };
static_assert(sizeof(V16) == 32, "V16 must be 32 bytes");

#define NWG 64
#define TPB 256
#define NW  (NWG * TPB / 32)   // 512 waves

__device__ __forceinline__ f32x8 wmma_bf(const V16& a, const V16& b, f32x8 c) {
  return __builtin_amdgcn_wmma_f32_16x16x32_bf16(false, a.v, false, b.v, (short)0, c, false, false);
}

// One 32(M) x 16(N) tile over K range [ks,ke), K multiple of 32, branchless.
// aload(row, k) returns 8 consecutive bf16 (16B) of A[row][k..k+7].
// Bt is [N][K] bf16 (transposed weights), row stride ldb. Accumulates c0/c1.
template <typename F>
__device__ __forceinline__ void gemm_tile(F&& aload, const __bf16* __restrict__ Bt,
                                          long ldb, int n0, int ks, int ke, int lane,
                                          f32x8& c0, f32x8& c1) {
  const int r0    = lane & 15;
  const int koffA = (lane & 16) ? 8  : 0;   // A half-wave K offset (ISA 7.12.2)
  const int koffB = (lane & 16) ? 8  : 0;   // B half-wave K offset (same as A)
  const __bf16* bp = Bt + (long)(n0 + r0) * ldb;
#pragma unroll 2
  for (int k = ks; k < ke; k += 32) {
    V16 fa0, fa1, fb;
    const int ka = k + koffA;
    fa0.q[0] = aload(r0, ka);
    fa0.q[1] = aload(r0, ka + 16);
    fa1.q[0] = aload(r0 + 16, ka);
    fa1.q[1] = aload(r0 + 16, ka + 16);
    const int kb = k + koffB;
    fb.q[0] = *(const u32x4*)(bp + kb);
    fb.q[1] = *(const u32x4*)(bp + kb + 16);
    c0 = wmma_bf(fa0, fb, c0);
    c1 = wmma_bf(fa1, fb, c1);
  }
}

__device__ __forceinline__ void store_tile(float* __restrict__ y, int ldy, int n0,
                                           int lane, const f32x8& c0, const f32x8& c1,
                                           float bv) {
  const int col = n0 + (lane & 15);
  const int rb  = (lane >> 4) * 8;           // C layout: vgpr r -> M = r + 8*(lane>=16)
#pragma unroll
  for (int i = 0; i < 8; i++) y[(rb + i) * ldy + col] = c0[i] + bv;
#pragma unroll
  for (int i = 0; i < 8; i++) y[(rb + 16 + i) * ldy + col] = c1[i] + bv;
}

// Masked store: y[row] = m_row * c + bias  (row mask = !is_first[row][t]).
__device__ __forceinline__ void store_tile_m(float* __restrict__ y, int ldy, int n0,
                                             int lane, const f32x8& c0, const f32x8& c1,
                                             float bv, const int* __restrict__ isf_t) {
  const int col = n0 + (lane & 15);
  const int rb  = (lane >> 4) * 8;
#pragma unroll
  for (int i = 0; i < 8; i++) {
    const float m = (isf_t[(rb + i) * 64] == 0) ? 1.f : 0.f;
    y[(rb + i) * ldy + col] = m * c0[i] + bv;
  }
#pragma unroll
  for (int i = 0; i < 8; i++) {
    const float m = (isf_t[(rb + 16 + i) * 64] == 0) ? 1.f : 0.f;
    y[(rb + 16 + i) * ldy + col] = m * c1[i] + bv;
  }
}

// Fused RMSNorm (+optional second partial buffer) + SiLU -> bf16, one row/wave.
__device__ __forceinline__ void act_row(const float* __restrict__ y,
                                        const float* __restrict__ y2,
                                        const float* __restrict__ g,
                                        __bf16* __restrict__ o, int n, int lane) {
  float ss = 0.f;
  for (int c = lane; c < n; c += 32) { float v = y[c] + (y2 ? y2[c] : 0.f); ss += v * v; }
  for (int off = 16; off; off >>= 1) ss += __shfl_xor(ss, off, 32);
  const float rs = rsqrtf(ss / (float)n + 1e-6f);
  for (int c = lane; c < n; c += 32) {
    float v = (y[c] + (y2 ? y2[c] : 0.f)) * rs * g[c];
    o[c] = (__bf16)(v / (1.f + expf(-v)));
  }
}

struct Params {
  // fp32 originals still needed in-loop
  const float *W_in_stoch, *b_in_stoch, *W_in_act, *b_in_act;
  const float *b_in_deter, *g_in_deter, *g_in_stoch, *g_in_act;
  const float *b_dyn, *g_dyn, *b_gru;
  const float *b_q0, *g_q0, *b_qo;
  const float *b_p0, *g_p0, *b_p1, *g_p1, *b_po;
  const int*   isf;                 // is_first [B][T]
  // bf16 transposed weights [N][K]
  const __bf16 *Wt0, *Wtdyn, *Wtgru, *Wtq0, *Wtqo, *Wtp0, *Wtp1, *Wtpo;
  const __bf16* embbf;              // [B][T][1024] bf16
  const float*  aeff;               // act/max(|act|,1)*mask, fp32 [B][T][16]
  // state / activations
  float*  deterf;   __bf16* deterbf;  __bf16* dbig;   // dbig: [T*32][2048] bf16
  __bf16 *xbf, *hbf, *xqbf, *xbig;
  float  *yx, *yxB, *yh, *yhB, *yq0, *yq0B, *gates, *ybig;
  float  *out_post, *out_prior, *out_deters;
  int*    cnt;
};

struct PrepParams {
  const float *W_in_deter, *W_dyn, *W_gru, *W_q0, *W_qo, *W_p0, *W_p1, *W_po;
  const float *embed, *action;
  const int*   isf;
  __bf16 *Wt0, *Wtdyn, *Wtgru, *Wtq0, *Wtqo, *Wtp0, *Wtp1, *Wtpo, *embbf, *deterbf;
  float  *aeff, *deterf;
  int*    cnt;
};

__device__ __forceinline__ void gridbar(int* cnt, int gen) {
  __threadfence();            // drain this thread's stores to device scope
  __syncthreads();
  if (threadIdx.x == 0) {
    __hip_atomic_fetch_add(cnt, 1, __ATOMIC_RELAXED, __HIP_MEMORY_SCOPE_AGENT);
    while (__hip_atomic_load(cnt, __ATOMIC_RELAXED, __HIP_MEMORY_SCOPE_AGENT) < NWG * gen)
      __builtin_amdgcn_s_sleep(1);
  }
  __syncthreads();
  __threadfence();            // acquire: invalidate per-CU L0 on every CU
}

// ------------------------- prep: convert / transpose / reset -----------------
__global__ void rssm_prep(PrepParams p) {
  const long N0 = 1024L * 2048;    // Wt0
  const long N1 = 8L * 256 * 3328; // Wtdyn
  const long N2 = 8L * 768 * 256;  // Wtgru
  const long N3 = 1024L * 3072;    // Wtq0
  const long N4 = 1024L * 1024;    // Wtqo
  const long N5 = 1024L * 2048;    // Wtp0
  const long N6 = 1024L * 1024;    // Wtp1
  const long N7 = 1024L * 1024;    // Wtpo
  const long N8 = 32L * 64 * 1024; // embbf
  const long N9 = 32L * 64 * 16;   // aeff
  const long N10 = 32L * 2048;     // zero state
  const long total = N0 + N1 + N2 + N3 + N4 + N5 + N6 + N7 + N8 + N9 + N10;
  for (long i = blockIdx.x * (long)blockDim.x + threadIdx.x; i < total;
       i += (long)gridDim.x * blockDim.x) {
    long r = i;
    if (r < N0)              { long n = r / 2048, k = r % 2048; p.Wt0[r]  = (__bf16)p.W_in_deter[k * 1024 + n]; }
    else if ((r -= N0) < N1) { long b = r / (256L * 3328), rem = r % (256L * 3328);
                               long n = rem / 3328, k = rem % 3328;
                               p.Wtdyn[r] = (__bf16)p.W_dyn[(b * 3328 + k) * 256 + n]; }
    else if ((r -= N1) < N2) { long b = r / (768L * 256), rem = r % (768L * 256);
                               long n = rem / 256, k = rem % 256;
                               p.Wtgru[r] = (__bf16)p.W_gru[(b * 256 + k) * 768 + n]; }
    else if ((r -= N2) < N3) { long n = r / 3072, k = r % 3072; p.Wtq0[r] = (__bf16)p.W_q0[k * 1024 + n]; }
    else if ((r -= N3) < N4) { long n = r / 1024, k = r % 1024; p.Wtqo[r] = (__bf16)p.W_qo[k * 1024 + n]; }
    else if ((r -= N4) < N5) { long n = r / 2048, k = r % 2048; p.Wtp0[r] = (__bf16)p.W_p0[k * 1024 + n]; }
    else if ((r -= N5) < N6) { long n = r / 1024, k = r % 1024; p.Wtp1[r] = (__bf16)p.W_p1[k * 1024 + n]; }
    else if ((r -= N6) < N7) { long n = r / 1024, k = r % 1024; p.Wtpo[r] = (__bf16)p.W_po[k * 1024 + n]; }
    else if ((r -= N7) < N8) { p.embbf[r] = (__bf16)p.embed[r]; }
    else if ((r -= N8) < N9) { long bt = r / 16; float a = p.action[r];
                               float m = (p.isf[bt] == 0) ? 1.f : 0.f;
                               p.aeff[r] = m * a / fmaxf(fabsf(a), 1.f); }
    else                     { r -= N9; p.deterf[r] = 0.f; p.deterbf[r] = (__bf16)0.f; }
  }
  if (blockIdx.x == 0 && threadIdx.x == 0) *p.cnt = 0;
}

// ------------------------------ persistent scan ------------------------------
__global__ __launch_bounds__(TPB, 1) void rssm_scan(Params p) {
  const int lane = threadIdx.x & 31;
  const int wid  = (blockIdx.x * TPB + threadIdx.x) >> 5;
  int gen = 0;
  const f32x8 Z = {0.f, 0.f, 0.f, 0.f, 0.f, 0.f, 0.f, 0.f};

  for (int t = 0; t < 64; ++t) {
    const int* isf_t = p.isf + t;     // is_first[b][t], row stride 64

    // ---- P1: in_deter GEMM (split-K=2) + stoch row-gather + act matvec ----
    for (int j = wid; j < 160; j += NW) {
      if (j < 128) {
        const int nt = j & 63, ks = j >> 6;
        f32x8 c0 = Z, c1 = Z;
        const __bf16* A = p.deterbf;
        auto ld = [&](int row, int kk) { return *(const u32x4*)(A + row * 2048 + kk); };
        gemm_tile(ld, p.Wt0, 2048, nt * 16, ks * 1024, ks * 1024 + 1024, lane, c0, c1);
        if (ks == 0) store_tile_m(p.yx, 3072, nt * 16, lane, c0, c1,
                                  p.b_in_deter[nt * 16 + (lane & 15)], isf_t);
        else         store_tile_m(p.yxB, 1024, nt * 16, lane, c0, c1, 0.f, isf_t);
      } else {
        const int b = j - 128;
        const bool m = (isf_t[b * 64] == 0);
        float acc[32];
#pragma unroll
        for (int c = 0; c < 32; c++) acc[c] = 0.f;
        if (t > 0 && m) {
          // argmax(post logits[b,t-1]) per class-group (group = lane)
          const float* lg = p.out_post + (long)(b * 64 + (t - 1)) * 1024 + lane * 32;
          float best = lg[0]; int bi = 0;
          for (int i2 = 1; i2 < 32; i2++) { float v = lg[i2]; if (v > best) { best = v; bi = i2; } }
          // stoch @ W_in_stoch == sum of selected rows of W (exact fp32)
          for (int g2 = 0; g2 < 32; g2++) {
            const int ig = __shfl(bi, g2, 32);
            const float* wr = p.W_in_stoch + (g2 * 32 + ig) * 1024;
#pragma unroll 4
            for (int c = 0; c < 32; c++) acc[c] += wr[lane + c * 32];
          }
        }
        float* yrow = p.yx + b * 3072;
        for (int c = 0; c < 32; c++) { int n = lane + c * 32; yrow[1024 + n] = acc[c] + p.b_in_stoch[n]; }
        // act head: K=16 fp32 matvec (a_eff has norm+mask baked in)
        const float* ap = p.aeff + (b * 64 + t) * 16;
        float av[16];
#pragma unroll
        for (int k2 = 0; k2 < 16; k2++) av[k2] = ap[k2];
        for (int c = 0; c < 32; c++) {
          int n = lane + c * 32;
          float s = p.b_in_act[n];
#pragma unroll
          for (int k2 = 0; k2 < 16; k2++) s += av[k2] * p.W_in_act[k2 * 1024 + n];
          yrow[2048 + n] = s;
        }
      }
    }
    gridbar(p.cnt, ++gen);

    // ---- P2: x = silu(rms(.)) for the 3 segments ----
    for (int j = wid; j < 96; j += NW) {
      const int seg = j >> 5, b = j & 31;
      const float* y  = p.yx + b * 3072 + seg * 1024;
      const float* y2 = (seg == 0) ? (p.yxB + b * 1024) : nullptr;
      const float* g  = (seg == 0) ? p.g_in_deter : (seg == 1) ? p.g_in_stoch : p.g_in_act;
      act_row(y, y2, g, p.xbf + b * 3072 + seg * 1024, 1024, lane);
    }
    gridbar(p.cnt, ++gen);

    // ---- P3: block-diagonal dyn GEMM, A = [deter_blk | x], split-K=2 ----
    // Region-pure calls: no pointer select inside the K loop.
    for (int j = wid; j < 256; j += NW) {
      const int blk = j >> 5, rem = j & 31, nt = rem & 15, ks = rem >> 4;
      f32x8 c0 = Z, c1 = Z;
      const __bf16* D  = p.deterbf + blk * 256;    // deter block, kk in [0,256)
      const __bf16* X  = p.xbf - 256;              // x region,    kk in [256,3328)
      auto ldD = [&](int row, int kk) { return *(const u32x4*)(D + row * 2048 + kk); };
      auto ldX = [&](int row, int kk) { return *(const u32x4*)(X + row * 3072 + kk); };
      const __bf16* Bt = p.Wtdyn + (long)blk * 256 * 3328;
      if (ks == 0) {
        gemm_tile(ldD, Bt, 3328, nt * 16, 0,   256,  lane, c0, c1);
        gemm_tile(ldX, Bt, 3328, nt * 16, 256, 1664, lane, c0, c1);
        store_tile_m(p.yh, 2048, blk * 256 + nt * 16, lane, c0, c1,
                     p.b_dyn[blk * 256 + nt * 16 + (lane & 15)], isf_t);
      } else {
        gemm_tile(ldX, Bt, 3328, nt * 16, 1664, 3328, lane, c0, c1);
        store_tile(p.yhB, 2048, blk * 256 + nt * 16, lane, c0, c1, 0.f); // x rows already masked
      }
    }
    gridbar(p.cnt, ++gen);

    // ---- P4: h = silu(rms(yh)) ----
    for (int j = wid; j < 32; j += NW)
      act_row(p.yh + j * 2048, p.yhB + j * 2048, p.g_dyn, p.hbf + j * 2048, 2048, lane);
    gridbar(p.cnt, ++gen);

    // ---- P5: per-block GRU GEMM + fused elementwise update (1 WG/block) ----
    if (blockIdx.x < 8) {
      const int blk = blockIdx.x, ww = threadIdx.x >> 5;
      for (int nt = ww; nt < 48; nt += 8) {
        f32x8 c0 = Z, c1 = Z;
        auto ld = [&](int row, int kk) { return *(const u32x4*)(p.hbf + row * 2048 + blk * 256 + kk); };
        gemm_tile(ld, p.Wtgru + (long)blk * 768 * 256, 256, nt * 16, 0, 256, lane, c0, c1);
        store_tile(p.gates + blk * 32 * 768, 768, nt * 16, lane, c0, c1,
                   p.b_gru[blk * 768 + nt * 16 + (lane & 15)]);
      }
      __threadfence();
      __syncthreads();
      const float* G = p.gates + blk * 32 * 768;
      for (int e = threadIdx.x; e < 32 * 256; e += TPB) {
        const int row = e >> 8, jc = e & 255;
        const float rg = 1.f / (1.f + expf(-G[row * 768 + jc]));
        const float cc = tanhf(rg * G[row * 768 + 256 + jc]);
        const float u  = 1.f / (1.f + expf(-(G[row * 768 + 512 + jc] - 1.f)));
        const float dg = (isf_t[row * 64] == 0) ? p.deterf[row * 2048 + blk * 256 + jc] : 0.f;
        const float nd = u * cc + (1.f - u) * dg;
        const int col = blk * 256 + jc;
        p.deterf[row * 2048 + col] = nd;
        const __bf16 nb = (__bf16)nd;
        p.deterbf[row * 2048 + col] = nb;
        p.dbig[(long)(t * 32 + row) * 2048 + col] = nb;
        p.out_deters[(long)(row * 64 + t) * 2048 + col] = nd;
      }
    }
    gridbar(p.cnt, ++gen);

    // ---- Q1: q0 GEMM, A = [deter | emb_t], split-K=2, region-pure calls ----
    for (int j = wid; j < 128; j += NW) {
      const int nt = j & 63, ks = j >> 6;
      f32x8 c0 = Z, c1 = Z;
      auto ldD = [&](int row, int kk) { return *(const u32x4*)(p.deterbf + row * 2048 + kk); };
      auto ldE = [&](int row, int kk) {
        return *(const u32x4*)(p.embbf + (long)(row * 64 + t) * 1024 + (kk - 2048));
      };
      if (ks == 0) {
        gemm_tile(ldD, p.Wtq0, 3072, nt * 16, 0, 1536, lane, c0, c1);
        store_tile(p.yq0, 1024, nt * 16, lane, c0, c1, p.b_q0[nt * 16 + (lane & 15)]);
      } else {
        gemm_tile(ldD, p.Wtq0, 3072, nt * 16, 1536, 2048, lane, c0, c1);
        gemm_tile(ldE, p.Wtq0, 3072, nt * 16, 2048, 3072, lane, c0, c1);
        store_tile(p.yq0B, 1024, nt * 16, lane, c0, c1, 0.f);
      }
    }
    gridbar(p.cnt, ++gen);

    // ---- Q2: xq = silu(rms(yq0)) ----
    for (int j = wid; j < 32; j += NW)
      act_row(p.yq0 + j * 1024, p.yq0B + j * 1024, p.g_q0, p.xqbf + j * 1024, 1024, lane);
    gridbar(p.cnt, ++gen);

    // ---- Q3: qo GEMM -> post logits (written straight to d_out) ----
    for (int j = wid; j < 64; j += NW) {
      f32x8 c0 = Z, c1 = Z;
      auto ld = [&](int row, int kk) { return *(const u32x4*)(p.xqbf + row * 1024 + kk); };
      gemm_tile(ld, p.Wtqo, 1024, j * 16, 0, 1024, lane, c0, c1);
      const int col = j * 16 + (lane & 15);
      const float bv = p.b_qo[col];
      const int rb = (lane >> 4) * 8;
#pragma unroll
      for (int i2 = 0; i2 < 8; i2++) p.out_post[(long)((rb + i2) * 64 + t) * 1024 + col] = c0[i2] + bv;
#pragma unroll
      for (int i2 = 0; i2 < 8; i2++) p.out_post[(long)((rb + 16 + i2) * 64 + t) * 1024 + col] = c1[i2] + bv;
    }
    gridbar(p.cnt, ++gen);
  }

  // ---- Epilogue: prior head batched over all B*T = 2048 rows ----
  for (int j = wid; j < 4096; j += NW) {                       // p0
    const int rp = j >> 6, nt = j & 63;
    f32x8 c0 = Z, c1 = Z;
    const __bf16* A = p.dbig + (long)rp * 32 * 2048;
    auto ld = [&](int row, int kk) { return *(const u32x4*)(A + (long)row * 2048 + kk); };
    gemm_tile(ld, p.Wtp0, 2048, nt * 16, 0, 2048, lane, c0, c1);
    store_tile(p.ybig + (long)rp * 32 * 1024, 1024, nt * 16, lane, c0, c1, p.b_p0[nt * 16 + (lane & 15)]);
  }
  gridbar(p.cnt, ++gen);
  for (int j = wid; j < 2048; j += NW)
    act_row(p.ybig + (long)j * 1024, nullptr, p.g_p0, p.xbig + (long)j * 1024, 1024, lane);
  gridbar(p.cnt, ++gen);
  for (int j = wid; j < 4096; j += NW) {                       // p1
    const int rp = j >> 6, nt = j & 63;
    f32x8 c0 = Z, c1 = Z;
    const __bf16* A = p.xbig + (long)rp * 32 * 1024;
    auto ld = [&](int row, int kk) { return *(const u32x4*)(A + row * 1024 + kk); };
    gemm_tile(ld, p.Wtp1, 1024, nt * 16, 0, 1024, lane, c0, c1);
    store_tile(p.ybig + (long)rp * 32 * 1024, 1024, nt * 16, lane, c0, c1, p.b_p1[nt * 16 + (lane & 15)]);
  }
  gridbar(p.cnt, ++gen);
  for (int j = wid; j < 2048; j += NW)
    act_row(p.ybig + (long)j * 1024, nullptr, p.g_p1, p.xbig + (long)j * 1024, 1024, lane);
  gridbar(p.cnt, ++gen);
  for (int j = wid; j < 4096; j += NW) {                       // po -> prior logits
    const int rp = j >> 6, nt = j & 63;
    f32x8 c0 = Z, c1 = Z;
    const __bf16* A = p.xbig + (long)rp * 32 * 1024;
    auto ld = [&](int row, int kk) { return *(const u32x4*)(A + row * 1024 + kk); };
    gemm_tile(ld, p.Wtpo, 1024, nt * 16, 0, 1024, lane, c0, c1);
    const int col = nt * 16 + (lane & 15);
    const float bv = p.b_po[col];
    const int rb = (lane >> 4) * 8;
#pragma unroll
    for (int i2 = 0; i2 < 8; i2++) { int r = rp * 32 + rb + i2;
      p.out_prior[(long)((r & 31) * 64 + (r >> 5)) * 1024 + col] = c0[i2] + bv; }
#pragma unroll
    for (int i2 = 0; i2 < 8; i2++) { int r = rp * 32 + 16 + rb + i2;
      p.out_prior[(long)((r & 31) * 64 + (r >> 5)) * 1024 + col] = c1[i2] + bv; }
  }
}

// ------------------------------- host side -----------------------------------
extern "C" void kernel_launch(void* const* d_in, const int* in_sizes, int n_in,
                              void* d_out, int out_size, void* d_ws, size_t ws_size,
                              hipStream_t stream) {
  (void)in_sizes; (void)n_in; (void)out_size; (void)ws_size;
  const float* embed      = (const float*)d_in[0];
  const float* action     = (const float*)d_in[1];
  const int*   isf        = (const int*)  d_in[2];
  const float* W_in_deter = (const float*)d_in[3];
  const float* b_in_deter = (const float*)d_in[4];
  const float* g_in_deter = (const float*)d_in[5];
  const float* W_in_stoch = (const float*)d_in[6];
  const float* b_in_stoch = (const float*)d_in[7];
  const float* g_in_stoch = (const float*)d_in[8];
  const float* W_in_act   = (const float*)d_in[9];
  const float* b_in_act   = (const float*)d_in[10];
  const float* g_in_act   = (const float*)d_in[11];
  const float* W_dyn      = (const float*)d_in[12];
  const float* b_dyn      = (const float*)d_in[13];
  const float* g_dyn      = (const float*)d_in[14];
  const float* W_gru      = (const float*)d_in[15];
  const float* b_gru      = (const float*)d_in[16];
  const float* W_p0       = (const float*)d_in[17];
  const float* b_p0       = (const float*)d_in[18];
  const float* g_p0       = (const float*)d_in[19];
  const float* W_p1       = (const float*)d_in[20];
  const float* b_p1       = (const float*)d_in[21];
  const float* g_p1       = (const float*)d_in[22];
  const float* W_po       = (const float*)d_in[23];
  const float* b_po       = (const float*)d_in[24];
  const float* W_q0       = (const float*)d_in[25];
  const float* b_q0       = (const float*)d_in[26];
  const float* g_q0       = (const float*)d_in[27];
  const float* W_qo       = (const float*)d_in[28];
  const float* b_qo       = (const float*)d_in[29];

  char* w = (char*)d_ws;
  auto alloc = [&](size_t bytes) { void* r = (void*)w; w += (bytes + 255) & ~(size_t)255; return r; };
  __bf16* Wt0    = (__bf16*)alloc(1024L * 2048 * 2);
  __bf16* Wtdyn  = (__bf16*)alloc(8L * 256 * 3328 * 2);
  __bf16* Wtgru  = (__bf16*)alloc(8L * 768 * 256 * 2);
  __bf16* Wtq0   = (__bf16*)alloc(1024L * 3072 * 2);
  __bf16* Wtqo   = (__bf16*)alloc(1024L * 1024 * 2);
  __bf16* Wtp0   = (__bf16*)alloc(1024L * 2048 * 2);
  __bf16* Wtp1   = (__bf16*)alloc(1024L * 1024 * 2);
  __bf16* Wtpo   = (__bf16*)alloc(1024L * 1024 * 2);
  __bf16* embbf  = (__bf16*)alloc(32L * 64 * 1024 * 2);
  float*  aeff   = (float*) alloc(32L * 64 * 16 * 4);
  float*  deterf = (float*) alloc(32L * 2048 * 4);
  __bf16* deterbf= (__bf16*)alloc(32L * 2048 * 2);
  __bf16* dbig   = (__bf16*)alloc(2048L * 2048 * 2);
  __bf16* xbf    = (__bf16*)alloc(32L * 3072 * 2);
  __bf16* hbf    = (__bf16*)alloc(32L * 2048 * 2);
  __bf16* xqbf   = (__bf16*)alloc(32L * 1024 * 2);
  __bf16* xbig   = (__bf16*)alloc(2048L * 1024 * 2);
  float*  yx     = (float*) alloc(32L * 3072 * 4);
  float*  yxB    = (float*) alloc(32L * 1024 * 4);
  float*  yh     = (float*) alloc(32L * 2048 * 4);
  float*  yhB    = (float*) alloc(32L * 2048 * 4);
  float*  yq0    = (float*) alloc(32L * 1024 * 4);
  float*  yq0B   = (float*) alloc(32L * 1024 * 4);
  float*  gates  = (float*) alloc(8L * 32 * 768 * 4);
  float*  ybig   = (float*) alloc(2048L * 1024 * 4);
  int*    cnt    = (int*)   alloc(256);

  PrepParams pp;
  pp.W_in_deter = W_in_deter; pp.W_dyn = W_dyn; pp.W_gru = W_gru; pp.W_q0 = W_q0;
  pp.W_qo = W_qo; pp.W_p0 = W_p0; pp.W_p1 = W_p1; pp.W_po = W_po;
  pp.embed = embed; pp.action = action; pp.isf = isf;
  pp.Wt0 = Wt0; pp.Wtdyn = Wtdyn; pp.Wtgru = Wtgru; pp.Wtq0 = Wtq0; pp.Wtqo = Wtqo;
  pp.Wtp0 = Wtp0; pp.Wtp1 = Wtp1; pp.Wtpo = Wtpo; pp.embbf = embbf; pp.deterbf = deterbf;
  pp.aeff = aeff; pp.deterf = deterf; pp.cnt = cnt;

  Params p;
  p.W_in_stoch = W_in_stoch; p.b_in_stoch = b_in_stoch; p.W_in_act = W_in_act; p.b_in_act = b_in_act;
  p.b_in_deter = b_in_deter; p.g_in_deter = g_in_deter; p.g_in_stoch = g_in_stoch; p.g_in_act = g_in_act;
  p.b_dyn = b_dyn; p.g_dyn = g_dyn; p.b_gru = b_gru;
  p.b_q0 = b_q0; p.g_q0 = g_q0; p.b_qo = b_qo;
  p.b_p0 = b_p0; p.g_p0 = g_p0; p.b_p1 = b_p1; p.g_p1 = g_p1; p.b_po = b_po;
  p.isf = isf;
  p.Wt0 = Wt0; p.Wtdyn = Wtdyn; p.Wtgru = Wtgru; p.Wtq0 = Wtq0; p.Wtqo = Wtqo;
  p.Wtp0 = Wtp0; p.Wtp1 = Wtp1; p.Wtpo = Wtpo; p.embbf = embbf; p.aeff = aeff;
  p.deterf = deterf; p.deterbf = deterbf; p.dbig = dbig;
  p.xbf = xbf; p.hbf = hbf; p.xqbf = xqbf; p.xbig = xbig;
  p.yx = yx; p.yxB = yxB; p.yh = yh; p.yhB = yhB; p.yq0 = yq0; p.yq0B = yq0B;
  p.gates = gates; p.ybig = ybig;
  p.out_post   = (float*)d_out;
  p.out_prior  = (float*)d_out + 2097152;   // B*T*32*32
  p.out_deters = (float*)d_out + 4194304;   // + another B*T*32*32
  p.cnt = cnt;

  rssm_prep<<<2048, 256, 0, stream>>>(pp);
  rssm_scan<<<NWG, TPB, 0, stream>>>(p);
}
